// MultiHeadAttentionLayer_41085657153550
// MI455X (gfx1250) — matrix operand order repaired and checked by reading, not compiled, not run on hardware
//
#include <hip/hip_runtime.h>

// ---------------------------------------------------------------------------
// MHA for MI455X (gfx1250, wave32). bf16 WMMA (v_wmma_f32_16x16x32_bf16) with
// f32 accumulate everywhere; flash attention with transposed-logits trick;
// GLOBAL_LOAD_ASYNC_TO_LDS_B128 + double buffering for K/V streaming;
// double-buffered LDS in the projection GEMMs.
// ---------------------------------------------------------------------------

#define B_DIM 2
#define S_DIM 2048
#define D_MODEL 1024
#define NUM_HEADS 16
#define DEPTH 64
#define NEG_BIG (-1e9f)

using u16 = unsigned short;
using u32 = unsigned int;
using u64 = unsigned long long;

typedef __attribute__((ext_vector_type(16))) __bf16 v16bf;
typedef __attribute__((ext_vector_type(8)))  float  v8f;

struct Frag { u32 u[8]; };                    // 16 bf16 = 8 dwords = 32 bytes
static_assert(sizeof(Frag) == sizeof(v16bf), "frag size");

__device__ __forceinline__ v16bf bfv(const Frag& f) {
  return __builtin_bit_cast(v16bf, f);
}

__device__ __forceinline__ u16 f2bf(float x) {        // RNE f32 -> bf16
  u32 u = __builtin_bit_cast(u32, x);
  u32 r = u + 0x7FFFu + ((u >> 16) & 1u);
  return (u16)(r >> 16);
}
__device__ __forceinline__ u32 pack2(float a, float b) {
  return (u32)f2bf(a) | ((u32)f2bf(b) << 16);
}

// --- CDNA5 async global->LDS (ASYNCcnt-tracked, ISA ch.8) -------------------
// saddr form: mem_addr = SGPR64 + VGPR32 + ioffset; LDS addr from vdst VGPR.
__device__ __forceinline__ void async_b128(u32 lds_addr, u32 voff, u64 sbase) {
  asm volatile("global_load_async_to_lds_b128 %0, %1, %2"
               :: "v"(lds_addr), "v"(voff), "s"(sbase)
               : "memory");
}
__device__ __forceinline__ void wait_async0() {
#if defined(__has_builtin)
#if __has_builtin(__builtin_amdgcn_s_wait_asynccnt)
  __builtin_amdgcn_s_wait_asynccnt(0);
#else
  asm volatile("s_wait_asynccnt 0" ::: "memory");
#endif
#else
  asm volatile("s_wait_asynccnt 0" ::: "memory");
#endif
}
__device__ __forceinline__ u32 lds_off(const void* p) {   // flat[31:0] == LDS offset
  return (u32)(u64)p;
}

// ---------------------------------------------------------------------------
// GEMM: out[M,N] = A[M,1024] * W[1024,N] + bias,  M=4096, N=K=1024.
// A_IS_F32: A fp32 (raw inputs) else bf16 (attention context).
// OUTMODE 0: fp32 row-major [4096,1024]        (final projection -> d_out)
// OUTMODE 1: bf16 [B,H,S,64]                   (Q, K for attention)
// OUTMODE 2: bf16 [B,H,64,S]  (transposed)     (V for attention)
// Block 256 thr (8 waves), tile BM=128 BN=64 BK=32, double-buffered LDS:
// global loads for step n+1 issue before the WMMAs of step n.
// ---------------------------------------------------------------------------
template<bool A_IS_F32, int OUTMODE>
__global__ __launch_bounds__(256) void wmma_gemm(const void* __restrict__ Ain,
                                                 const float* __restrict__ W,
                                                 const float* __restrict__ bias,
                                                 void* __restrict__ out)
{
  constexpr int Kdim = D_MODEL, N = D_MODEL;
  __shared__ __align__(16) u16 Asm[2][128][40];   // 32 K + 8 pad (80B rows)
  __shared__ __align__(16) u16 Bsm[2][64][40];    // transposed [n][k]

  const int t    = threadIdx.x;
  const int n0   = blockIdx.x * 64;
  const int m0   = blockIdx.y * 128;
  const int lane = t & 31, w = t >> 5;
  const int q    = lane & 15, hi = lane >> 4;

  const int arow = t >> 1;           // 0..127
  const int acb  = (t & 1) * 16;     // 0 or 16
  const int bk   = t >> 3;           // 0..31
  const int bnb  = (t & 7) * 8;      // 0..56

  u16 aR[16], bR[8];
  auto gload = [&](int k0) {
    if constexpr (A_IS_F32) {
      const float* ap = (const float*)Ain + (size_t)(m0 + arow) * Kdim + k0 + acb;
#pragma unroll
      for (int i = 0; i < 16; ++i) aR[i] = f2bf(ap[i]);
    } else {
      const u16* ap = (const u16*)Ain + (size_t)(m0 + arow) * Kdim + k0 + acb;
#pragma unroll
      for (int i = 0; i < 16; ++i) aR[i] = ap[i];
    }
    const float* wp = W + (size_t)(k0 + bk) * N + n0 + bnb;
#pragma unroll
    for (int i = 0; i < 8; ++i) bR[i] = f2bf(wp[i]);
  };
  auto lstore = [&](int buf) {
#pragma unroll
    for (int i = 0; i < 16; ++i) Asm[buf][arow][acb + i] = aR[i];
#pragma unroll
    for (int i = 0; i < 8; ++i) Bsm[buf][bnb + i][bk] = bR[i];
  };

  v8f acc[4] = {};

  gload(0);
  lstore(0);
  __syncthreads();

  for (int k0 = 0; k0 < Kdim; k0 += 32) {
    const int buf = (k0 >> 5) & 1;
    const bool pf = (k0 + 32 < Kdim);
    if (pf) gload(k0 + 32);                 // global loads in flight over WMMAs

    Frag af;
    {
      const u32* ar = (const u32*)&Asm[buf][w * 16 + q][0];
#pragma unroll
      for (int v = 0; v < 8; ++v) {
        const int kk = 16 * hi + 2 * v;
        af.u[v] = ar[kk >> 1];
      }
    }
#pragma unroll
    for (int nt = 0; nt < 4; ++nt) {
      Frag bf;
      const u32* br = (const u32*)&Bsm[buf][nt * 16 + q][0];
#pragma unroll
      for (int v = 0; v < 8; ++v) bf.u[v] = br[8 * hi + v];   // K = 16*hi + 2v + j
      acc[nt] = __builtin_amdgcn_wmma_f32_16x16x32_bf16(
          false, bfv(af), false, bfv(bf), (short)0, acc[nt], false, false);
    }

    if (pf) lstore(buf ^ 1);                // write other buffer, then barrier
    __syncthreads();
  }

  // ---- epilogue: C layout lane holds col n = l&15, rows m = r + 8*hi ----
#pragma unroll
  for (int nt = 0; nt < 4; ++nt) {
    const int n = n0 + nt * 16 + q;
    const float bv = bias[n];
#pragma unroll
    for (int r = 0; r < 8; ++r) {
      const int m = m0 + w * 16 + r + 8 * hi;
      const float val = acc[nt][r] + bv;
      if constexpr (OUTMODE == 0) {
        ((float*)out)[(size_t)m * N + n] = val;
      } else if constexpr (OUTMODE == 1) {
        const int h = n >> 6, d = n & 63;
        const int b = m >> 11, s = m & (S_DIM - 1);
        ((u16*)out)[(((size_t)b * NUM_HEADS + h) * S_DIM + s) * DEPTH + d] = f2bf(val);
      } else {                              // V transposed: [B,H,64,S]
        const int h = n >> 6, d = n & 63;
        const int b = m >> 11, s = m & (S_DIM - 1);
        ((u16*)out)[(((size_t)b * NUM_HEADS + h) * DEPTH + d) * S_DIM + s] = f2bf(val);
      }
    }
  }
}

// ---------------------------------------------------------------------------
// Flash attention. Q,K: bf16 [B,H,S,64]; V: bf16 [B,H,64,S] (pre-transposed);
// ctx: bf16 [B,S,H*64]. Block = 8 waves = 128 q-rows of one (b,h).
// K and V^T chunks (32 keys) stream via global_load_async_to_lds_b128 into
// double-buffered LDS; prefetch of chunk n+1 overlaps compute of chunk n.
// Logits computed transposed (S^T = K_tile x Q^T): softmax over keys is
// per-lane + one shfl_xor(16); S^T C-layout == P A-layout (transpose free).
// ---------------------------------------------------------------------------
__global__ __launch_bounds__(256) void attn_kernel(const u16* __restrict__ Q,
                                                   const u16* __restrict__ Kt,
                                                   const u16* __restrict__ Vt,
                                                   const int* __restrict__ mask,
                                                   u16* __restrict__ ctx)
{
  __shared__ float maskAdd[S_DIM];                // 8 KB
  __shared__ __align__(16) u16 Ksm[2][32][72];    // [key][d], 144B rows (16B-mult)
  __shared__ __align__(16) u16 Vsm[2][64][40];    // [d][key],  80B rows (16B-mult)

  const int t    = threadIdx.x;
  const int bh   = blockIdx.y;
  const int b    = bh >> 4;
  const int h    = bh & 15;
  const int lane = t & 31, w = t >> 5;
  const int q    = lane & 15, hi = lane >> 4;
  const float scale = 0.125f;                     // 1/sqrt(64)

  for (int i = t; i < S_DIM; i += 256)
    maskAdd[i] = NEG_BIG * (float)mask[b * S_DIM + i];

  // Q^T B-fragments (held all pass): B[d][q], col q = l&15, K(d) = 16*hi+2v+j
  const int sq = blockIdx.x * 128 + w * 16 + q;
  Frag qb0, qb1;
  {
    const u32* qr = (const u32*)(Q + ((size_t)bh * S_DIM + sq) * DEPTH);
#pragma unroll
    for (int v = 0; v < 8; ++v) {
      qb0.u[v] = qr[8 * hi + v];
      qb1.u[v] = qr[16 + 8 * hi + v];
    }
  }

  // async staging maps: K chunk 32x64 (16B/thread), V^T chunk 64x32 (16B/thread)
  const int kk  = t >> 3;            // K row   0..31
  const int db  = (t & 7) * 8;       // K depth 0..56
  const int vd  = t >> 2;            // V^T row (d) 0..63
  const int vkb = (t & 3) * 8;       // V^T key 0,8,16,24
  const u64 kbase = (u64)(const void*)(Kt + (size_t)bh * S_DIM * DEPTH);
  const u64 vbase = (u64)(const void*)(Vt + (size_t)bh * DEPTH * S_DIM);

  auto stage = [&](int kc, int buf) {
    async_b128(lds_off(&Ksm[buf][kk][db]), (u32)(((kc + kk) * DEPTH + db) * 2), kbase);
    async_b128(lds_off(&Vsm[buf][vd][vkb]), (u32)((vd * S_DIM + kc + vkb) * 2), vbase);
  };

  v8f o[4] = {};
  float m_run = -3.0e38f, l_run = 0.0f;

  stage(0, 0);

  for (int kc = 0; kc < S_DIM; kc += 32) {
    wait_async0();                       // own async stores to LDS landed
    __syncthreads();                     // everyone's landed; prev buf free
    const int buf = (kc >> 5) & 1;
    if (kc + 32 < S_DIM) stage(kc + 32, buf ^ 1);   // prefetch over compute

    // ---- K A-fragments: 2 key-subtiles x 2 depth-halves ----
    Frag ka[2][2];
#pragma unroll
    for (int kt = 0; kt < 2; ++kt) {
      const u32* kr = (const u32*)&Ksm[buf][kt * 16 + q][0];
#pragma unroll
      for (int v = 0; v < 8; ++v) {
        const int d0 = 16 * hi + 2 * v;
        ka[kt][0].u[v] = kr[d0 >> 1];
        ka[kt][1].u[v] = kr[(d0 + 32) >> 1];
      }
    }
    // ---- V B-fragments: 4 depth-subtiles from V^T rows (dword reads) ----
    Frag vb[4];
#pragma unroll
    for (int nt = 0; nt < 4; ++nt) {
      const u32* vr = (const u32*)&Vsm[buf][nt * 16 + q][0];
#pragma unroll
      for (int v = 0; v < 8; ++v) vb[nt].u[v] = vr[(v < 4) ? (4 * hi + v) : (8 + 4 * hi + (v - 4))];
    }

    // ---- logits transposed: S^T = K_tile(16k x 64d) x Q^T(64d x 16q) ----
    v8f s0 = {}, s1 = {};
    s0 = __builtin_amdgcn_wmma_f32_16x16x32_bf16(false, bfv(ka[0][0]), false, bfv(qb0), (short)0, s0, false, false);
    s0 = __builtin_amdgcn_wmma_f32_16x16x32_bf16(false, bfv(ka[0][1]), false, bfv(qb1), (short)0, s0, false, false);
    s1 = __builtin_amdgcn_wmma_f32_16x16x32_bf16(false, bfv(ka[1][0]), false, bfv(qb0), (short)0, s1, false, false);
    s1 = __builtin_amdgcn_wmma_f32_16x16x32_bf16(false, bfv(ka[1][1]), false, bfv(qb1), (short)0, s1, false, false);

    // ---- scale + mask; online softmax over the 32-key chunk ----
    float p0[8], p1[8];
    float mloc = -3.0e38f;
#pragma unroll
    for (int r = 0; r < 8; ++r) {
      const float a = s0[r] * scale + maskAdd[kc + r + 8 * hi];
      const float c = s1[r] * scale + maskAdd[kc + 16 + r + 8 * hi];
      p0[r] = a; p1[r] = c;
      mloc = fmaxf(mloc, fmaxf(a, c));
    }
    mloc = fmaxf(mloc, __shfl_xor(mloc, 16, 32));   // lanes l, l^16 share q
    const float m_new  = fmaxf(m_run, mloc);
    const float factor = __expf(m_run - m_new);
    float lloc = 0.0f;
#pragma unroll
    for (int r = 0; r < 8; ++r) {
      p0[r] = __expf(p0[r] - m_new); lloc += p0[r];
      p1[r] = __expf(p1[r] - m_new); lloc += p1[r];
    }
    lloc  = lloc + __shfl_xor(lloc, 16, 32);
    l_run = l_run * factor + lloc;
    m_run = m_new;

    // ---- pack P: S^T C-layout == P A-layout, bf16-pack in order ----
    Frag pa;
#pragma unroll
    for (int v = 0; v < 8; ++v)
      pa.u[v] = (v < 4) ? pack2(p0[2 * v], p0[2 * v + 1])
                        : pack2(p1[2 * (v - 4)], p1[2 * (v - 4) + 1]);

    // ---- rescale O rows (factor for q-row r+8*hi lives in lane r+8*hi) ----
#pragma unroll
    for (int r = 0; r < 8; ++r) {
      const float fr = __shfl(factor, r + 8 * hi, 32);
      o[0][r] *= fr; o[1][r] *= fr; o[2][r] *= fr; o[3][r] *= fr;
    }
    // ---- O += P x V ----
#pragma unroll
    for (int nt = 0; nt < 4; ++nt)
      o[nt] = __builtin_amdgcn_wmma_f32_16x16x32_bf16(
          false, bfv(pa), false, bfv(vb[nt]), (short)0, o[nt], false, false);
  }

  // ---- normalize and write context bf16 [B,S,H*64] (merged heads) ----
#pragma unroll
  for (int r = 0; r < 8; ++r) {
    const float lr  = __shfl(l_run, r + 8 * hi, 32);
    const float inv = 1.0f / lr;
    const int srow  = blockIdx.x * 128 + w * 16 + r + 8 * hi;
    const size_t base = ((size_t)b * S_DIM + srow) * D_MODEL + h * DEPTH;
#pragma unroll
    for (int nt = 0; nt < 4; ++nt)
      ctx[base + nt * 16 + q] = f2bf(o[nt][r] * inv);
  }
}

// ---------------------------------------------------------------------------
// Launcher. Inputs (setup_inputs order):
//  0 query 1 key 2 value 3 mask 4 Wq 5 bq 6 Wk 7 bk 8 Wv 9 bv 10 Wo 11 bo
// Workspace: Q,K bf16 [B,H,S,64], V bf16 [B,H,64,S], ctx bf16 [4096,1024].
// ---------------------------------------------------------------------------
extern "C" void kernel_launch(void* const* d_in, const int* in_sizes, int n_in,
                              void* d_out, int out_size, void* d_ws, size_t ws_size,
                              hipStream_t stream) {
  (void)in_sizes; (void)n_in; (void)out_size; (void)ws_size;

  const float* query = (const float*)d_in[0];
  const float* key   = (const float*)d_in[1];
  const float* value = (const float*)d_in[2];
  const int*   mask  = (const int*)d_in[3];
  const float* Wq = (const float*)d_in[4];  const float* bq = (const float*)d_in[5];
  const float* Wk = (const float*)d_in[6];  const float* bk = (const float*)d_in[7];
  const float* Wv = (const float*)d_in[8];  const float* bv = (const float*)d_in[9];
  const float* Wo = (const float*)d_in[10]; const float* bo = (const float*)d_in[11];

  const size_t elems = (size_t)B_DIM * NUM_HEADS * S_DIM * DEPTH;  // 4M bf16
  u16* wsQ = (u16*)d_ws;
  u16* wsK = wsQ + elems;
  u16* wsV = wsK + elems;
  u16* wsC = wsV + elems;

  dim3 blk(256);
  dim3 ggemm(D_MODEL / 64, (B_DIM * S_DIM) / 128);   // (16, 32)
  dim3 gattn(S_DIM / 128, B_DIM * NUM_HEADS);        // (16, 32)

  wmma_gemm<true, 1><<<ggemm, blk, 0, stream>>>(query, Wq, bq, wsQ);
  wmma_gemm<true, 1><<<ggemm, blk, 0, stream>>>(key,   Wk, bk, wsK);
  wmma_gemm<true, 2><<<ggemm, blk, 0, stream>>>(value, Wv, bv, wsV);
  attn_kernel<<<gattn, blk, 0, stream>>>(wsQ, wsK, wsV, mask, wsC);
  wmma_gemm<false, 0><<<ggemm, blk, 0, stream>>>(wsC, Wo, bo, d_out);
}
